// VisionAttention_18167711662193
// MI455X (gfx1250) — compile-verified, matrix operand order checked by reading, not run on hardware
//
#include <hip/hip_runtime.h>

// ---------------------------------------------------------------------------
// CDNA5 / gfx1250 vision attention: bf16 WMMA (v_wmma_f32_16x16x32_bf16)
//   S=16384, D=1280, 64 segments x 256 tokens, 16 heads, head_dim=80
// Round 2: GEMM A-tile fill via GLOBAL_LOAD_ASYNC_TO_LDS_B128 (ASYNCcnt path)
// ---------------------------------------------------------------------------

typedef __attribute__((ext_vector_type(16))) __bf16       v16bf;
typedef __attribute__((ext_vector_type(8)))  float        v8f;
typedef __attribute__((ext_vector_type(4)))  unsigned int u32x4;

union Bf16Frag { u32x4 u[2]; v16bf v; };

#define S_TOT   16384
#define D_DIM   1280
#define QKV_LD  3840
#define N_SEG   64
#define SEG_L   256
#define N_HEAD  16
#define HEAD_D  80
#define SCALE_F 0.111803398874989485f   // 80^-0.5

static __device__ __forceinline__ unsigned short f32_to_bf16_rne(float f) {
    union { float f; unsigned int u; } x; x.f = f;
    unsigned int u = x.u;
    u += 0x7FFFu + ((u >> 16) & 1u);          // round-to-nearest-even
    return (unsigned short)(u >> 16);
}

// Async copy 16B global -> LDS (no VGPR round trip, tracked by ASYNCcnt).
// LDS address comes from the vdst VGPR plus offset; generic shared ptr low 32 bits
// are the wave-relative LDS byte offset (LDS aperture lives in addr[63:32]).
static __device__ __forceinline__ void async_g2l_b128(void* lds_dst,
                                                      const void* gsrc) {
    unsigned lds_off = (unsigned)(unsigned long long)lds_dst;
    unsigned long long ga = (unsigned long long)gsrc;
    asm volatile("global_load_async_to_lds_b128 %0, %1, off"
                 :: "v"(lds_off), "v"(ga) : "memory");
}
static __device__ __forceinline__ void wait_asynccnt0() {
    asm volatile("s_wait_asynccnt 0x0" ::: "memory");
}

// Load a 16x32 bf16 A/B fragment from row-major storage (LDS or global).
// Lane %16 = row (A) / col (B); lane/16 selects K-halves {0..7,16..23} vs
// {8..15,24..31} per ISA 7.12.2.  Two aligned 16B chunks.
static __device__ __forceinline__ v16bf load_frag(const unsigned short* base,
                                                  int ld, int lane, int kbase) {
    const int r    = lane & 15;
    const int half = lane >> 4;
    const unsigned short* p = base + r * ld + kbase + half * 8;
    Bf16Frag f;
    f.u[0] = *(const u32x4*)(p);
    f.u[1] = *(const u32x4*)(p + 16);
    return f.v;
}

// Same, but from a per-lane row pointer with zero-padding past head_dim=80.
static __device__ __forceinline__ v16bf load_frag_pad80(const unsigned short* rowp,
                                                        int half, int kbase) {
    const int c0 = kbase + half * 8;
    const int c1 = c0 + 16;
    u32x4 z = {0u, 0u, 0u, 0u};
    Bf16Frag f;
    f.u[0] = (c0 < HEAD_D) ? *(const u32x4*)(rowp + c0) : z;
    f.u[1] = (c1 < HEAD_D) ? *(const u32x4*)(rowp + c1) : z;
    return f.v;
}

// ---------------------------------------------------------------------------
// f32 -> bf16 conversion (grid-stride)
// ---------------------------------------------------------------------------
__global__ void cvt_f32_bf16(const float* __restrict__ in,
                             unsigned short* __restrict__ out, int n) {
    for (int i = blockIdx.x * blockDim.x + threadIdx.x; i < n;
         i += gridDim.x * blockDim.x)
        out[i] = f32_to_bf16_rne(in[i]);
}

// ---------------------------------------------------------------------------
// Tiled bf16 GEMM:  C[M,N] = A[M,K] @ B[K,N] + bias[N]
// Block tile 128x128, K-tile 64.  8 waves, each wave: 64x32 (4x2 WMMA tiles).
// A tile filled by async global->LDS DMA; B tile transposed through VGPRs.
// LDS rows padded to 72 bf16 (144 B = 9*16) -> aligned ds_load_b128 frags.
// ---------------------------------------------------------------------------
template <bool OUT_BF16>
__global__ __launch_bounds__(256)
void gemm_bf16(const unsigned short* __restrict__ A,
               const unsigned short* __restrict__ B,
               const float* __restrict__ bias,
               void* __restrict__ Cout, int M, int N, int K) {
    __shared__ unsigned short As[128][72];   // [m][k]
    __shared__ unsigned short Bs[128][72];   // [n][k]  (transposed on fill)

    const int tid  = threadIdx.x;
    const int lane = tid & 31;
    const int wid  = tid >> 5;
    const int wm   = wid >> 2;               // 0..1 -> 64 rows
    const int wn   = wid & 3;                // 0..3 -> 32 cols
    const int bm0  = blockIdx.y * 128;
    const int bn0  = blockIdx.x * 128;

    v8f acc[4][2];
    #pragma unroll
    for (int mt = 0; mt < 4; ++mt)
        #pragma unroll
        for (int nt = 0; nt < 2; ++nt)
            acc[mt][nt] = (v8f){0.f,0.f,0.f,0.f,0.f,0.f,0.f,0.f};

    for (int k0 = 0; k0 < K; k0 += 64) {
        __syncthreads();                     // WAR vs previous frag reads

        // A tile: 128x64, async 16B global->LDS, 4 DMAs/thread
        #pragma unroll
        for (int i = 0; i < 4; ++i) {
            int id  = tid + i * 256;         // 0..1023
            int row = id >> 3;
            int kc  = (id & 7) * 8;
            async_g2l_b128(&As[row][kc],
                           A + (size_t)(bm0 + row) * K + k0 + kc);
        }

        // Prefetch next B tile rows into GL2 while this tile is consumed.
        if (k0 + 64 < K) {
            int kr = tid >> 2;               // 0..63
            int nc = (tid & 3) * 32;
            __builtin_prefetch(B + (size_t)(k0 + 64 + kr) * N + bn0 + nc, 0, 1);
        }

        // B tile: 64x128, load 8 along N, transpose-scatter into Bs[n][k]
        #pragma unroll
        for (int i = 0; i < 4; ++i) {
            int id = tid + i * 256;
            int kr = id >> 4;                // 0..63
            int nc = (id & 15) * 8;          // 0..120
            u32x4 val = *(const u32x4*)(B + (size_t)(k0 + kr) * N + bn0 + nc);
            const unsigned short* pv = (const unsigned short*)&val;
            #pragma unroll
            for (int e = 0; e < 8; ++e) Bs[nc + e][kr] = pv[e];
        }

        wait_asynccnt0();                    // A-tile DMA complete
        __syncthreads();

        #pragma unroll
        for (int ks = 0; ks < 2; ++ks) {
            v16bf bfrag[2];
            #pragma unroll
            for (int nt = 0; nt < 2; ++nt)
                bfrag[nt] = load_frag(&Bs[wn * 32 + nt * 16][0], 72, lane, ks * 32);
            #pragma unroll
            for (int mt = 0; mt < 4; ++mt) {
                v16bf afrag = load_frag(&As[wm * 64 + mt * 16][0], 72, lane, ks * 32);
                #pragma unroll
                for (int nt = 0; nt < 2; ++nt)
                    acc[mt][nt] = __builtin_amdgcn_wmma_f32_16x16x32_bf16(
                        false, afrag, false, bfrag[nt],
                        (short)0, acc[mt][nt], false, false);
            }
        }
    }

    // Epilogue: D layout -> element (m = r + 8*half, n = lane%16) in VGPR r
    const int r16  = lane & 15;
    const int half = lane >> 4;
    #pragma unroll
    for (int mt = 0; mt < 4; ++mt)
        #pragma unroll
        for (int nt = 0; nt < 2; ++nt) {
            const int n = bn0 + wn * 32 + nt * 16 + r16;
            const float bv = bias[n];
            #pragma unroll
            for (int r = 0; r < 8; ++r) {
                const int m = bm0 + wm * 64 + mt * 16 + half * 8 + r;
                const float v = acc[mt][nt][r] + bv;
                if (OUT_BF16)
                    ((unsigned short*)Cout)[(size_t)m * N + n] = f32_to_bf16_rne(v);
                else
                    ((float*)Cout)[(size_t)m * N + n] = v;
            }
        }
}

// ---------------------------------------------------------------------------
// Fused attention: one workgroup (8 waves) per (segment, head).
// LDS: scores f32 [256][260] (266240 B) + vT bf16 [80][264] (42240 B)
//      = 308480 B  (fits in the 320 KB WGP LDS; one WG per WGP).
// Wave w owns query rows [32w, 32w+32): QK^T and softmax are barrier-free;
// softmax packs bf16 probabilities in place over its own f32 rows.
// One barrier (for vT) before P @ V.
// ---------------------------------------------------------------------------
#define SC_LD  260          // f32 words per score row
#define VT_LD  264          // bf16 per vT row
#define ATTN_SMEM (SEG_L * SC_LD * 4 + HEAD_D * VT_LD * 2)

__global__ __launch_bounds__(256)
void attn_fused(const unsigned short* __restrict__ qkv,
                unsigned short* __restrict__ attn_out) {
    extern __shared__ char smem[];
    float*          scores = (float*)smem;                       // [256][260]
    unsigned short* psm    = (unsigned short*)smem;              // bf16 view
    unsigned short* vT     = (unsigned short*)(smem + SEG_L * SC_LD * 4);

    const int head = blockIdx.x;
    const int seg0 = blockIdx.y * SEG_L;
    const int tid  = threadIdx.x;
    const int lane = tid & 31;
    const int wid  = tid >> 5;
    const int r16  = lane & 15;
    const int half = lane >> 4;
    const int m0   = wid * 32;              // this wave's 32 query rows

    // ---- Phase 0: V^T into LDS (thread t handles seq position t) ----------
    {
        const unsigned short* vp =
            qkv + (size_t)(seg0 + tid) * QKV_LD + 2 * D_DIM + head * HEAD_D;
        #pragma unroll 8
        for (int d = 0; d < HEAD_D; ++d) vT[d * VT_LD + tid] = vp[d];
    }

    // ---- Phase A: S = (Q K^T) * scale  (rows m0..m0+31, all 256 cols) -----
    v16bf aq[2][3];
    #pragma unroll
    for (int mt = 0; mt < 2; ++mt) {
        const unsigned short* qp =
            qkv + (size_t)(seg0 + m0 + mt * 16 + r16) * QKV_LD + head * HEAD_D;
        #pragma unroll
        for (int ks = 0; ks < 3; ++ks)
            aq[mt][ks] = load_frag_pad80(qp, half, ks * 32);
    }
    for (int nt = 0; nt < 16; ++nt) {
        v8f a0 = (v8f){0.f,0.f,0.f,0.f,0.f,0.f,0.f,0.f};
        v8f a1 = a0;
        const unsigned short* kp =
            qkv + (size_t)(seg0 + nt * 16 + r16) * QKV_LD + D_DIM + head * HEAD_D;
        #pragma unroll
        for (int ks = 0; ks < 3; ++ks) {
            v16bf bf = load_frag_pad80(kp, half, ks * 32);
            a0 = __builtin_amdgcn_wmma_f32_16x16x32_bf16(false, aq[0][ks], false, bf,
                                                         (short)0, a0, false, false);
            a1 = __builtin_amdgcn_wmma_f32_16x16x32_bf16(false, aq[1][ks], false, bf,
                                                         (short)0, a1, false, false);
        }
        #pragma unroll
        for (int r = 0; r < 8; ++r) {
            scores[(m0 +      half * 8 + r) * SC_LD + nt * 16 + r16] = a0[r] * SCALE_F;
            scores[(m0 + 16 + half * 8 + r) * SC_LD + nt * 16 + r16] = a1[r] * SCALE_F;
        }
    }

    // ---- Phase B: row softmax, pack bf16 probs in place (wave-private) ----
    for (int rr = 0; rr < 32; ++rr) {
        const int r = m0 + rr;
        float v[8];
        #pragma unroll
        for (int j = 0; j < 8; ++j) v[j] = scores[r * SC_LD + lane + 32 * j];
        float mx = v[0];
        #pragma unroll
        for (int j = 1; j < 8; ++j) mx = fmaxf(mx, v[j]);
        #pragma unroll
        for (int off = 16; off >= 1; off >>= 1) mx = fmaxf(mx, __shfl_xor(mx, off));
        float sum = 0.f;
        #pragma unroll
        for (int j = 0; j < 8; ++j) { v[j] = __expf(v[j] - mx); sum += v[j]; }
        #pragma unroll
        for (int off = 16; off >= 1; off >>= 1) sum += __shfl_xor(sum, off);
        const float inv = 1.f / sum;
        unsigned short* pr = psm + (size_t)r * (SC_LD * 2);
        #pragma unroll
        for (int j = 0; j < 8; ++j)
            pr[lane + 32 * j] = f32_to_bf16_rne(v[j] * inv);
    }

    __syncthreads();   // vT visible to all; probs are wave-private anyway

    // ---- Phase C: O = P @ V  (K = 256 -> 8 WMMA k-steps) ------------------
    #pragma unroll
    for (int mt = 0; mt < 2; ++mt) {
        v16bf ap[8];
        #pragma unroll
        for (int ks = 0; ks < 8; ++ks)
            ap[ks] = load_frag(psm + (size_t)(m0 + mt * 16) * (SC_LD * 2),
                               SC_LD * 2, lane, ks * 32);
        #pragma unroll
        for (int nt = 0; nt < 5; ++nt) {
            v8f acc = (v8f){0.f,0.f,0.f,0.f,0.f,0.f,0.f,0.f};
            #pragma unroll
            for (int ks = 0; ks < 8; ++ks) {
                v16bf bf = load_frag(vT + (size_t)(nt * 16) * VT_LD,
                                     VT_LD, lane, ks * 32);
                acc = __builtin_amdgcn_wmma_f32_16x16x32_bf16(
                    false, ap[ks], false, bf, (short)0, acc, false, false);
            }
            #pragma unroll
            for (int r = 0; r < 8; ++r) {
                const int m = seg0 + m0 + mt * 16 + half * 8 + r;
                const int c = head * HEAD_D + nt * 16 + r16;
                attn_out[(size_t)m * D_DIM + c] = f32_to_bf16_rne(acc[r]);
            }
        }
    }
}

// ---------------------------------------------------------------------------
// Host launch
// ---------------------------------------------------------------------------
extern "C" void kernel_launch(void* const* d_in, const int* in_sizes, int n_in,
                              void* d_out, int out_size, void* d_ws, size_t ws_size,
                              hipStream_t stream) {
    (void)in_sizes; (void)n_in; (void)out_size; (void)ws_size;
    const float* hidden = (const float*)d_in[0];
    // d_in[1] = cu_seqlens (uniform segments, unused), d_in[2] = num_heads (=16)
    const float* w_qkv  = (const float*)d_in[3];
    const float* b_qkv  = (const float*)d_in[4];
    const float* w_proj = (const float*)d_in[5];
    const float* b_proj = (const float*)d_in[6];
    float* out = (float*)d_out;

    char* ws = (char*)d_ws;
    unsigned short* hid_bf   = (unsigned short*)ws;                 ws += (size_t)S_TOT * D_DIM * 2;
    unsigned short* wqkv_bf  = (unsigned short*)ws;                 ws += (size_t)D_DIM * QKV_LD * 2;
    unsigned short* wproj_bf = (unsigned short*)ws;                 ws += (size_t)D_DIM * D_DIM * 2;
    unsigned short* qkv_bf   = (unsigned short*)ws;                 ws += (size_t)S_TOT * QKV_LD * 2;
    unsigned short* attn_bf  = (unsigned short*)ws;

    // 1) f32 -> bf16 converts
    cvt_f32_bf16<<<4096, 256, 0, stream>>>(hidden, hid_bf,   S_TOT * D_DIM);
    cvt_f32_bf16<<<2048, 256, 0, stream>>>(w_qkv,  wqkv_bf,  D_DIM * QKV_LD);
    cvt_f32_bf16<<<1024, 256, 0, stream>>>(w_proj, wproj_bf, D_DIM * D_DIM);

    // 2) QKV projection: [16384,1280] x [1280,3840] -> bf16 qkv (+bias)
    gemm_bf16<true><<<dim3(QKV_LD / 128, S_TOT / 128), 256, 0, stream>>>(
        hid_bf, wqkv_bf, b_qkv, qkv_bf, S_TOT, QKV_LD, D_DIM);

    // 3) Fused attention: one WG per (seg, head); 301 KB dynamic LDS
    hipFuncSetAttribute((const void*)attn_fused,
                        hipFuncAttributeMaxDynamicSharedMemorySize, ATTN_SMEM);
    attn_fused<<<dim3(N_HEAD, N_SEG), 256, ATTN_SMEM, stream>>>(qkv_bf, attn_bf);

    // 4) Output projection: [16384,1280] x [1280,1280] -> f32 out (+bias)
    gemm_bf16<false><<<dim3(D_DIM / 128, S_TOT / 128), 256, 0, stream>>>(
        attn_bf, wproj_bf, b_proj, out, S_TOT, D_DIM, D_DIM);
}
